// BA_GCN_71339406786966
// MI455X (gfx1250) — matrix operand order repaired and checked by reading, not compiled, not run on hardware
//
#include <hip/hip_runtime.h>
#include <hip/hip_bf16.h>

typedef __attribute__((ext_vector_type(16))) _Float16 v16h;
typedef __attribute__((ext_vector_type(8)))  _Float16 v8h;
typedef __attribute__((ext_vector_type(8)))  float    v8f;

#define HDIM  128          // F == H == 128
#define HCOLS 384          // [ root | W0 | W1 ] concatenated output columns
#define NBLK  64           // output columns per wave (4 x 16 WMMA tiles)
#define LWROW 136          // padded LDS row stride in halfs (272 B -> bank spread)
#define LDSBYTES (HCOLS * LWROW * 2)

// ---------------------------------------------------------------------------
// fp32 -> fp16 conversion (row-major, same layout)
// ---------------------------------------------------------------------------
__global__ void k_f32_to_f16(const float* __restrict__ in, _Float16* __restrict__ out,
                             long long n) {
  long long i = (long long)blockIdx.x * blockDim.x + threadIdx.x;
  if (i < n) out[i] = (_Float16)in[i];
}

// ---------------------------------------------------------------------------
// Pack [root(128x128) | W0(128x128) | W1(128x128)] -> wt[384][128] f16,
// TRANSPOSED so wt[j][k] = Wcat[k][j]; each output column owns a contiguous
// K-run, matching the per-lane B-operand layout.
// ---------------------------------------------------------------------------
__global__ void k_pack_weights(const float* __restrict__ root, const float* __restrict__ W,
                               _Float16* __restrict__ wt) {
  int idx = blockIdx.x * blockDim.x + threadIdx.x;   // 384*128
  if (idx >= HCOLS * HDIM) return;
  int j = idx >> 7;     // output column 0..383
  int k = idx & 127;    // K index
  float v;
  if (j < 128)      v = root[k * HDIM + j];
  else if (j < 256) v = W[0 * HDIM * HDIM + k * HDIM + (j - 128)];
  else              v = W[1 * HDIM * HDIM + k * HDIM + (j - 256)];
  wt[idx] = (_Float16)v;
}

// ---------------------------------------------------------------------------
// Fused GEMM: hcat[N][384] = xh[N][128] @ Wcat[128][384] via WMMA f16->f32.
//
// Workgroup stage 0: async-copy the whole packed weight (96 KB) into LDS with
// global_load_async_to_lds_b128 (ASYNCcnt -> s_wait_asynccnt), rows padded to
// 272 B for bank spreading. All 8 waves then feed B operands from LDS.
//
// Per wave: one 16x64 output tile, 4 f32 accumulators, 16 WMMAs; the A
// operand is reused across the 4 column tiles of every K-step.
//
// A (16x32 f16): lane l, m=l&15, half=l>>4,
//   elems 0..7  -> K = ks + 8*half + i ; elems 8..15 -> K = ks + 16 + 8*half + i
// B (32x16 f16): lane l, n=l&15, same K order as A
//   (wt pre-transposed -> two contiguous 16-byte runs per lane)
// D (16x16 f32): elem r -> row m = 8*half + r, col n = l&15.
// ---------------------------------------------------------------------------
__global__ void k_gemm_wmma(const _Float16* __restrict__ xh, const _Float16* __restrict__ wt,
                            float* __restrict__ hcat, int n_rows, long long total_tiles) {
  extern __shared__ _Float16 lwt[];          // [HCOLS][LWROW]

  // ---- stage packed weights into LDS (async copy, 16 B per lane per issue)
  {
    const int tid = threadIdx.x;
    const unsigned long long gbase = (unsigned long long)(uintptr_t)wt;
    const unsigned lbase = (unsigned)(uintptr_t)lwt;
#pragma unroll
    for (int it = 0; it < (HCOLS * 16) / 256; ++it) {   // 6144 chunks / 256 thr
      const int chunk = it * 256 + tid;
      const int row = chunk >> 4;                       // 0..383
      const int c16 = chunk & 15;                       // 16B chunk within row
      const unsigned laddr = lbase + (unsigned)(row * (LWROW * 2) + c16 * 16);
      const unsigned long long gaddr = gbase + (unsigned long long)(row * (HDIM * 2) + c16 * 16);
      asm volatile("global_load_async_to_lds_b128 %0, %1, off"
                   :: "v"(laddr), "v"(gaddr) : "memory");
    }
    asm volatile("s_wait_asynccnt 0x0" ::: "memory");
    __syncthreads();
  }

  const int wave = threadIdx.x >> 5;
  const int lane = threadIdx.x & 31;
  const long long tile = (long long)blockIdx.x * (blockDim.x >> 5) + wave;
  if (tile >= total_tiles) return;          // after barrier; uniform per wave
  const int ntn = HCOLS / NBLK;             // 6 column blocks
  const int tm  = (int)(tile / ntn);
  const int tnb = (int)(tile % ntn);
  const int half = lane >> 4;
  const int lx   = lane & 15;

  const bool full = (tm * 16 + 16 <= n_rows);
  int mrow = tm * 16 + lx;
  if (mrow >= n_rows) mrow = n_rows - 1;    // clamp loads (stores guarded in tail path)

  const _Float16* __restrict__ arow = xh + (size_t)mrow * HDIM;
  const _Float16* browl = lwt + (size_t)(tnb * NBLK + lx) * LWROW;

  v8f acc0 = {}, acc1 = {}, acc2 = {}, acc3 = {};
#pragma unroll
  for (int ks = 0; ks < HDIM; ks += 32) {
    const v8h a0 = *(const v8h*)(arow + ks + 8 * half);
    const v8h a1 = *(const v8h*)(arow + ks + 16 + 8 * half);
    v16h a;
#pragma unroll
    for (int i = 0; i < 8; ++i) { a[i] = a0[i]; a[8 + i] = a1[i]; }

    const int kb = ks + 8 * half;
#pragma unroll
    for (int j = 0; j < 4; ++j) {
      const _Float16* bp = browl + (size_t)(j * 16) * LWROW + kb;
      const v8h b0 = *(const v8h*)(bp);      // ds_load_b128
      const v8h b1 = *(const v8h*)(bp + 16); // ds_load_b128
      v16h b;
#pragma unroll
      for (int i = 0; i < 8; ++i) { b[i] = b0[i]; b[8 + i] = b1[i]; }
      v8f c = (j == 0) ? acc0 : (j == 1) ? acc1 : (j == 2) ? acc2 : acc3;
      c = __builtin_amdgcn_wmma_f32_16x16x32_f16(false, a, false, b, (short)0, c,
                                                 false, false);
      if (j == 0) acc0 = c; else if (j == 1) acc1 = c;
      else if (j == 2) acc2 = c; else acc3 = c;
    }
  }

  const int colb = tnb * NBLK + lx;
  float* __restrict__ orow = hcat + (size_t)(tm * 16 + (half << 3)) * HCOLS + colb;
  if (full) {
#pragma unroll
    for (int r = 0; r < 8; ++r) {
      float* p = orow + (size_t)r * HCOLS;
      p[0]  = acc0[r];
      p[16] = acc1[r];
      p[32] = acc2[r];
      p[48] = acc3[r];
    }
  } else {
#pragma unroll
    for (int r = 0; r < 8; ++r) {
      const int mo = tm * 16 + (half << 3) + r;
      if (mo < n_rows) {
        float* p = hcat + (size_t)mo * HCOLS + colb;
        p[0]  = acc0[r];
        p[16] = acc1[r];
        p[32] = acc2[r];
        p[48] = acc3[r];
      }
    }
  }
}

// ---------------------------------------------------------------------------
// Edge counts (layer-invariant): cnt[dst][r] += 1 per edge, then invert once.
// ---------------------------------------------------------------------------
__global__ void k_count(const int* __restrict__ dst, const int* __restrict__ etype,
                        float* __restrict__ cnt, long long E_) {
  long long e = (long long)blockIdx.x * blockDim.x + threadIdx.x;
  if (e >= E_) return;
  atomicAdd(cnt + (size_t)dst[e] * 2 + etype[e], 1.0f);
}

__global__ void k_invcnt(float* __restrict__ cnt, long long n) {
  long long i = (long long)blockIdx.x * blockDim.x + threadIdx.x;
  if (i < n) cnt[i] = 1.0f / fmaxf(cnt[i], 1.0f);
}

// ---------------------------------------------------------------------------
// Edge scatter: 32 lanes per edge; each lane gathers float4 from the edge's
// relation slice of hcat and atomically accumulates into num[dst][r][128].
// ---------------------------------------------------------------------------
__global__ void k_scatter(const float* __restrict__ hcat, const int* __restrict__ src,
                          const int* __restrict__ dst, const int* __restrict__ etype,
                          float* __restrict__ num, long long E_) {
  long long t = (long long)blockIdx.x * blockDim.x + threadIdx.x;
  long long e = t >> 5;
  if (e >= E_) return;
  const int c4 = (int)(t & 31) * 4;
  const int r = etype[e];
  const int s = src[e];
  const int d = dst[e];
  const float4 v = *(const float4*)(hcat + (size_t)s * HCOLS + HDIM + r * HDIM + c4);
  float* p = num + (size_t)d * (2 * HDIM) + r * HDIM + c4;
  atomicAdd(p + 0, v.x);
  atomicAdd(p + 1, v.y);
  atomicAdd(p + 2, v.z);
  atomicAdd(p + 3, v.w);
}

// ---------------------------------------------------------------------------
// Combine: out = root_part + b + num0*inv0 + num1*inv1; ReLU.
// Writes fp32 (for pooling) AND the f16 copy for the next layer's GEMM.
// ---------------------------------------------------------------------------
__global__ void k_combine(const float* __restrict__ hcat, const float* __restrict__ num,
                          const float* __restrict__ inv, const float* __restrict__ bias,
                          float* __restrict__ xout, _Float16* __restrict__ xh,
                          long long total) {
  long long i = (long long)blockIdx.x * blockDim.x + threadIdx.x;
  if (i >= total) return;
  long long n = i >> 7;
  int c = (int)(i & 127);
  float v = hcat[n * HCOLS + c] + bias[c]
          + num[n * (2 * HDIM) + c]        * inv[n * 2 + 0]
          + num[n * (2 * HDIM) + HDIM + c] * inv[n * 2 + 1];
  v = fmaxf(v, 0.0f);
  xout[i] = v;
  xh[i] = (_Float16)v;
}

// ---------------------------------------------------------------------------
// Global mean pool: atomic segment-sum into psum[G][128], pcnt[G].
// ---------------------------------------------------------------------------
__global__ void k_pool(const float* __restrict__ h, const int* __restrict__ batch,
                       float* __restrict__ psum, float* __restrict__ pcnt, long long total) {
  long long i = (long long)blockIdx.x * blockDim.x + threadIdx.x;
  if (i >= total) return;
  long long n = i >> 7;
  int c = (int)(i & 127);
  int g = batch[n];
  atomicAdd(psum + (size_t)g * HDIM + c, h[i]);
  if (c == 0) atomicAdd(pcnt + g, 1.0f);
}

// ---------------------------------------------------------------------------
// Final tiny linear: out[G][C] = (psum/max(pcnt,1)) @ lin_w[128][C] + lin_b.
// ---------------------------------------------------------------------------
__global__ void k_final(const float* __restrict__ psum, const float* __restrict__ pcnt,
                        const float* __restrict__ lin_w, const float* __restrict__ lin_b,
                        float* __restrict__ out, int G, int C) {
  int i = blockIdx.x * blockDim.x + threadIdx.x;
  if (i >= G * C) return;
  int g = i / C, cc = i % C;
  float inv = 1.0f / fmaxf(pcnt[g], 1.0f);
  float s = 0.0f;
#pragma unroll 8
  for (int k = 0; k < HDIM; ++k) s += psum[(size_t)g * HDIM + k] * lin_w[k * C + cc];
  out[i] = s * inv + lin_b[cc];
}

// ---------------------------------------------------------------------------
extern "C" void kernel_launch(void* const* d_in, const int* in_sizes, int n_in,
                              void* d_out, int out_size, void* d_ws, size_t ws_size,
                              hipStream_t stream) {
  const float* x     = (const float*)d_in[0];
  const int*   eidx  = (const int*)d_in[1];
  const int*   etype = (const int*)d_in[2];
  const int*   batch = (const int*)d_in[3];
  const float* W[3]    = {(const float*)d_in[4], (const float*)d_in[7],  (const float*)d_in[10]};
  const float* root[3] = {(const float*)d_in[5], (const float*)d_in[8],  (const float*)d_in[11]};
  const float* bvec[3] = {(const float*)d_in[6], (const float*)d_in[9],  (const float*)d_in[12]};
  const float* lin_w = (const float*)d_in[13];
  const float* lin_b = (const float*)d_in[14];

  const long long N = in_sizes[0] / HDIM;
  const long long E = in_sizes[2];
  const int C = in_sizes[14];
  const int G = out_size / C;

  // workspace carve-out (256B aligned slices)
  char* ws = (char*)d_ws;
  size_t off = 0;
  auto take = [&](size_t bytes) -> char* {
    char* p = ws + off;
    off = (off + bytes + 255) & ~(size_t)255;
    return p;
  };
  _Float16* xh  = (_Float16*)take((size_t)N * HDIM * sizeof(_Float16));
  _Float16* wt  = (_Float16*)take((size_t)HCOLS * HDIM * sizeof(_Float16));
  float* hcat   = (float*)take((size_t)N * HCOLS * sizeof(float));
  float* num    = (float*)take((size_t)N * 2 * HDIM * sizeof(float));
  float* inv    = (float*)take((size_t)N * 2 * sizeof(float));
  float* xbuf   = (float*)take((size_t)N * HDIM * sizeof(float));
  float* psum   = (float*)take((size_t)G * HDIM * sizeof(float));
  float* pcnt   = (float*)take((size_t)G * sizeof(float));
  (void)ws_size; (void)n_in;

  const int BT = 256;
  const int* srcp = eidx;
  const int* dstp = eidx + E;

  // initial fp32 -> fp16 conversion of node features
  {
    long long n = N * HDIM;
    k_f32_to_f16<<<(int)((n + BT - 1) / BT), BT, 0, stream>>>(x, xh, n);
  }

  // layer-invariant edge counts -> reciprocals (computed once)
  hipMemsetAsync(inv, 0, (size_t)N * 2 * sizeof(float), stream);
  k_count<<<(int)((E + BT - 1) / BT), BT, 0, stream>>>(dstp, etype, inv, E);
  {
    long long n = N * 2;
    k_invcnt<<<(int)((n + BT - 1) / BT), BT, 0, stream>>>(inv, n);
  }

  const long long tiles_m = (N + 15) / 16;
  const long long total_tiles = tiles_m * (HCOLS / NBLK);
  const int gemm_blocks = (int)((total_tiles + 7) / 8);     // 8 waves per block

  for (int l = 0; l < 3; ++l) {
    k_pack_weights<<<(HCOLS * HDIM + BT - 1) / BT, BT, 0, stream>>>(root[l], W[l], wt);
    k_gemm_wmma<<<gemm_blocks, BT, LDSBYTES, stream>>>(xh, wt, hcat, (int)N, total_tiles);
    hipMemsetAsync(num, 0, (size_t)N * 2 * HDIM * sizeof(float), stream);
    {
      long long sthreads = E * 32;
      k_scatter<<<(int)((sthreads + BT - 1) / BT), BT, 0, stream>>>(hcat, srcp, dstp, etype,
                                                                    num, E);
    }
    {
      long long n = N * HDIM;
      k_combine<<<(int)((n + BT - 1) / BT), BT, 0, stream>>>(hcat, num, inv, bvec[l],
                                                             xbuf, xh, n);
    }
  }

  hipMemsetAsync(psum, 0, (size_t)G * HDIM * sizeof(float), stream);
  hipMemsetAsync(pcnt, 0, (size_t)G * sizeof(float), stream);
  {
    long long n = N * HDIM;
    k_pool<<<(int)((n + BT - 1) / BT), BT, 0, stream>>>(xbuf, batch, psum, pcnt, n);
  }
  {
    int n = G * C;
    k_final<<<(n + BT - 1) / BT, BT, 0, stream>>>(psum, pcnt, lin_w, lin_b, (float*)d_out, G, C);
  }
}
